// TransformerModel_7155415515183
// MI455X (gfx1250) — matrix operand order repaired and checked by reading, not compiled, not run on hardware
//
#include <hip/hip_runtime.h>
#include <stdint.h>

// ---------------- model constants ----------------
#define NLAYERS 6
#define DMODEL  512
#define NHEAD   8
#define DHEAD   64
#define FFDIM   2048
#define VOCAB   32000
#define BATCH   8
#define SEQ     512
#define ROWS    (BATCH * SEQ)   // 4096 token rows

// ---------------- WMMA plumbing ------------------
typedef __attribute__((ext_vector_type(16))) __bf16         v16bf;
typedef __attribute__((ext_vector_type(8)))  float          v8f;
typedef __attribute__((ext_vector_type(8)))  unsigned short ushort8;

__device__ __forceinline__ unsigned short f2bf_bits(float f) {
  union { float f; unsigned u; } x; x.f = f;
  unsigned u = x.u;
  return (unsigned short)((u + 0x7FFFu + ((u >> 16) & 1u)) >> 16);  // RNE
}

union FragU { v16bf v; ushort8 h[2]; };

// A-matrix 16x32 bf16 fragment (ISA 7.12.2): lane<16 -> K {0..7,16..23},
// lane>=16 -> K {8..15,24..31}. p = base of this row's 32-wide K chunk.
__device__ __forceinline__ v16bf frag_a(const unsigned short* p, int half8) {
  FragU u;
  u.h[0] = *(const ushort8*)(p + half8);
  u.h[1] = *(const ushort8*)(p + half8 + 16);
  return u.v;
}
// B-matrix 32x16 bf16 fragment: lane<16 -> K {0..7,16..23}, lane>=16 -> K {8..15,24..31},
// column = lane&15. p = base of this column's contiguous K run.
__device__ __forceinline__ v16bf frag_b(const unsigned short* p, int half16) {
  FragU u;
  u.h[0] = *(const ushort8*)(p + (half16 >> 1));
  u.h[1] = *(const ushort8*)(p + (half16 >> 1) + 16);
  return u.v;
}
__device__ __forceinline__ v8f wmma_bf16(v16bf a, v16bf b, v8f c) {
  return __builtin_amdgcn_wmma_f32_16x16x32_bf16(false, a, false, b,
                                                 (short)0, c, false, false);
}

// CDNA5 async global->LDS copy (16B per lane), tracked by ASYNCcnt.
__device__ __forceinline__ void async_copy16(const unsigned short* gptr,
                                             unsigned short* lptr) {
  unsigned lds = (unsigned)(uintptr_t)lptr;         // low 32 bits = LDS offset
  unsigned long long ga = (unsigned long long)(uintptr_t)gptr;
  asm volatile("global_load_async_to_lds_b128 %0, %1, off"
               :: "v"(lds), "v"(ga) : "memory");
}
__device__ __forceinline__ void wait_async0() {
  asm volatile("s_wait_asynccnt 0" ::: "memory");
}

// ---------------- weight convert + transpose: W[K][N] f32 -> Wt[N][K] bf16 ----------------
__global__ __launch_bounds__(256) void transpose_cvt_kernel(
    const float* __restrict__ W, unsigned short* __restrict__ Wt,
    int K, int N) {
  __shared__ unsigned short t[32][33];
  int tx = threadIdx.x, ty = threadIdx.y;           // 32 x 8
  int n0 = blockIdx.x * 32, k0 = blockIdx.y * 32;
  #pragma unroll
  for (int i = 0; i < 4; ++i) {
    int k = ty + i * 8;
    t[tx][k] = f2bf_bits(W[(size_t)(k0 + k) * N + n0 + tx]);
  }
  __syncthreads();
  #pragma unroll
  for (int i = 0; i < 4; ++i) {
    int n = ty + i * 8;
    Wt[(size_t)(n0 + n) * K + k0 + tx] = t[n][tx];
  }
}

// ---------------- embedding + positional encoding (f32 + bf16 copies) ----------------
__global__ __launch_bounds__(256) void embed_kernel(
    const int* __restrict__ tok, const float* __restrict__ emb,
    const float* __restrict__ pe, float* __restrict__ out,
    unsigned short* __restrict__ outb) {
  int bs = blockIdx.x;
  int s  = bs & (SEQ - 1);
  int t  = tok[bs];
  size_t ob = (size_t)bs * DMODEL;
  size_t eb = (size_t)t * DMODEL;
  size_t pb = (size_t)s * DMODEL;
  #pragma unroll
  for (int i = 0; i < 2; ++i) {
    int d = threadIdx.x + i * 256;
    float v = emb[eb + d] + pe[pb + d];
    out[ob + d]  = v;
    outb[ob + d] = f2bf_bits(v);
  }
}

// ---------------- GEMM: C[M,N] = A[M,K]bf16 @ Wt[N,K]bf16^T + bias ----------------
// LDS-free: fragments load directly from global (weights are L2-resident on a
// 192MB-L2 part). 256 threads = 8 waves, block tile 64M x 256N, wave 32M x 64N
// (2x4 WMMA tiles), K-step 32. Requires M%64==0, N%256==0, K%32==0.
__global__ __launch_bounds__(256) void gemm_bf16_kernel(
    const unsigned short* __restrict__ A,    // [M][K] bf16
    const unsigned short* __restrict__ Wt,   // [N][K] bf16
    const float* __restrict__ bias,
    float* __restrict__ Cf, unsigned short* __restrict__ Cb,
    int M, int N, int K, int relu) {
  const int lane = threadIdx.x & 31;
  const int wave = threadIdx.x >> 5;         // 0..7
  const int wm = (wave >> 2) * 32;           // 0,32
  const int wn = (wave & 3) * 64;            // 0,64,128,192
  const int m0 = blockIdx.y * 64;
  const int n0 = blockIdx.x * 256;
  const int r15   = lane & 15;
  const int ahalf = (lane < 16) ? 0 : 8;
  const int bhalf = (lane < 16) ? 0 : 16;
  const int mhalf = (lane < 16) ? 0 : 8;

  const unsigned short* arow[2];
  #pragma unroll
  for (int i = 0; i < 2; ++i)
    arow[i] = A + (size_t)(m0 + wm + i * 16 + r15) * K;
  const unsigned short* bcol[4];
  #pragma unroll
  for (int j = 0; j < 4; ++j)
    bcol[j] = Wt + (size_t)(n0 + wn + j * 16 + r15) * K;

  v8f acc[2][4];
  #pragma unroll
  for (int i = 0; i < 2; ++i)
    #pragma unroll
    for (int j = 0; j < 4; ++j)
      acc[i][j] = (v8f){0.f,0.f,0.f,0.f,0.f,0.f,0.f,0.f};

  for (int k0 = 0; k0 < K; k0 += 32) {
    __builtin_prefetch(bcol[0] + k0 + 256, 0, 1);   // global_prefetch_b8
    __builtin_prefetch(arow[0] + k0 + 256, 0, 1);
    v16bf a[2], b[4];
    #pragma unroll
    for (int i = 0; i < 2; ++i) a[i] = frag_a(arow[i] + k0, ahalf);
    #pragma unroll
    for (int j = 0; j < 4; ++j) b[j] = frag_b(bcol[j] + k0, bhalf);
    #pragma unroll
    for (int i = 0; i < 2; ++i)
      #pragma unroll
      for (int j = 0; j < 4; ++j)
        acc[i][j] = wmma_bf16(a[i], b[j], acc[i][j]);
  }

  #pragma unroll
  for (int i = 0; i < 2; ++i)
    #pragma unroll
    for (int j = 0; j < 4; ++j) {
      int col = n0 + wn + j * 16 + r15;
      float bv = bias[col];
      #pragma unroll
      for (int e = 0; e < 8; ++e) {
        int row = m0 + wm + i * 16 + mhalf + e;
        float v = acc[i][j][e] + bv;
        if (relu) v = fmaxf(v, 0.f);
        if (Cf) Cf[(size_t)row * N + col] = v;
        if (Cb) Cb[(size_t)row * N + col] = f2bf_bits(v);
      }
    }
}

// ---------------- fused flash-attention, per (b,h), bf16 in / bf16 out ----------------
// Q/K/V: bf16 [B,H,S,64] contiguous (== reference's direct reshape of [B,S,D]).
// Out:   bf16 [B,Sq,DMODEL]; head h writes columns [h*64, h*64+64).
#define QSTR 72   // 64 + pad (144B rows, 16B aligned)
#define PSTR 40   // 32 + pad (80B rows, 16B aligned)

__global__ __launch_bounds__(128) void attention_kernel(
    const unsigned short* __restrict__ Q, const unsigned short* __restrict__ K,
    const unsigned short* __restrict__ V, unsigned short* __restrict__ Out,
    const int* __restrict__ mask, const int* __restrict__ pad,
    int Sq, int Sk) {
  __shared__ alignas(16) unsigned short Qs[64 * QSTR];       // [q][dh]
  __shared__ alignas(16) unsigned short Ks[32 * QSTR];       // [key][dh]
  __shared__ alignas(16) unsigned short Vt[64 * PSTR];       // [dh][key]
  __shared__ alignas(16) unsigned short Ps[4 * 16 * PSTR];   // per-wave P tile

  const int tid  = threadIdx.x;
  const int lane = tid & 31;
  const int wave = tid >> 5;
  const int bh = blockIdx.y;
  const int b  = bh / NHEAD;
  const int h  = bh - b * NHEAD;
  const int q0 = blockIdx.x * 64;
  const int qr0 = wave * 16;

  const size_t qbase = (size_t)bh * Sq * DHEAD;
  const size_t kbase = (size_t)bh * Sk * DHEAD;

  const int r15   = lane & 15;
  const int nl    = lane & 15;
  const int ahalf = (lane < 16) ? 0 : 8;
  const int bhalf = (lane < 16) ? 0 : 16;
  const int mhalf = (lane < 16) ? 0 : 8;

  // async-copy Q tile (64 x 64 bf16, 16B per lane per op)
  #pragma unroll
  for (int i = 0; i < 4; ++i) {
    int chunk = i * 128 + tid;                // 512 chunks of 8 shorts
    int r = chunk >> 3, c = (chunk & 7) * 8;
    async_copy16(Q + qbase + (size_t)(q0 + r) * DHEAD + c, &Qs[r * QSTR + c]);
  }
  wait_async0();
  __syncthreads();

  float m_i[8], l_i[8];
  v8f Of[4];
  #pragma unroll
  for (int e = 0; e < 8; ++e) { m_i[e] = -1e30f; l_i[e] = 0.f; }
  #pragma unroll
  for (int t = 0; t < 4; ++t) Of[t] = (v8f){0.f,0.f,0.f,0.f,0.f,0.f,0.f,0.f};

  unsigned short* Psw = &Ps[wave * 16 * PSTR];

  for (int kblk = 0; kblk < Sk; kblk += 32) {
    // async-copy K block rows; transpose V block scalar (bf16 moves, no cvt)
    #pragma unroll
    for (int i = 0; i < 2; ++i) {
      int chunk = i * 128 + tid;              // 256 chunks
      int r = chunk >> 3, c = (chunk & 7) * 8;
      async_copy16(K + kbase + (size_t)(kblk + r) * DHEAD + c, &Ks[r * QSTR + c]);
    }
    #pragma unroll
    for (int i = 0; i < 16; ++i) {
      int idx = i * 128 + tid;
      int r = idx >> 6, c = idx & 63;
      Vt[c * PSTR + r] = V[kbase + (size_t)(kblk + r) * DHEAD + c];
    }
    wait_async0();
    __syncthreads();

    // logits: Q(16x64) . K^T(64x32) -> two 16x16 tiles
    v8f L0 = (v8f){0.f,0.f,0.f,0.f,0.f,0.f,0.f,0.f};
    v8f L1 = L0;
    #pragma unroll
    for (int c = 0; c < 2; ++c) {
      v16bf aq  = frag_a(&Qs[(qr0 + r15) * QSTR + c * 32], ahalf);
      v16bf bk0 = frag_b(&Ks[(r15) * QSTR + c * 32], bhalf);
      v16bf bk1 = frag_b(&Ks[(16 + r15) * QSTR + c * 32], bhalf);
      L0 = wmma_bf16(aq, bk0, L0);
      L1 = wmma_bf16(aq, bk1, L1);
    }

    // scale + masks (-inf modeled as -1e30: exp underflows to 0, no NaN)
    #pragma unroll
    for (int e = 0; e < 8; ++e) {
      int qg = q0 + qr0 + mhalf + e;
      int kg0 = kblk + nl, kg1 = kblk + 16 + nl;
      float lv0 = L0[e] * 0.125f, lv1 = L1[e] * 0.125f;   // 1/sqrt(64)
      if (pad) {
        if (pad[(size_t)b * Sk + kg0] != 0) lv0 = -1e30f;
        if (pad[(size_t)b * Sk + kg1] != 0) lv1 = -1e30f;
      }
      if (mask) {
        if (mask[(size_t)qg * Sk + kg0] != 0) lv0 = -1e30f;
        if (mask[(size_t)qg * Sk + kg1] != 0) lv1 = -1e30f;
      }
      L0[e] = lv0; L1[e] = lv1;
    }

    // online softmax (row lives in 16 lanes of one half-wave)
    #pragma unroll
    for (int e = 0; e < 8; ++e) {
      float rm = fmaxf(L0[e], L1[e]);
      rm = fmaxf(rm, __shfl_xor(rm, 1, 32));
      rm = fmaxf(rm, __shfl_xor(rm, 2, 32));
      rm = fmaxf(rm, __shfl_xor(rm, 4, 32));
      rm = fmaxf(rm, __shfl_xor(rm, 8, 32));
      float mnew = fmaxf(m_i[e], rm);
      float corr = __expf(m_i[e] - mnew);
      m_i[e] = mnew;
      l_i[e] *= corr;
      Of[0][e] *= corr; Of[1][e] *= corr; Of[2][e] *= corr; Of[3][e] *= corr;
      float p0 = 0.f, p1 = 0.f;
      if (mnew > -1e29f) { p0 = __expf(L0[e] - mnew); p1 = __expf(L1[e] - mnew); }
      float ps = p0 + p1;
      ps += __shfl_xor(ps, 1, 32);
      ps += __shfl_xor(ps, 2, 32);
      ps += __shfl_xor(ps, 4, 32);
      ps += __shfl_xor(ps, 8, 32);
      l_i[e] += ps;
      Psw[(mhalf + e) * PSTR + nl]      = f2bf_bits(p0);
      Psw[(mhalf + e) * PSTR + 16 + nl] = f2bf_bits(p1);
    }

    // O += P(16x32) @ V(32x64): 4 n-tiles
    v16bf ap = frag_a(&Psw[r15 * PSTR], ahalf);
    #pragma unroll
    for (int t = 0; t < 4; ++t) {
      v16bf bv = frag_b(&Vt[(t * 16 + r15) * PSTR], bhalf);
      Of[t] = wmma_bf16(ap, bv, Of[t]);
    }
    __syncthreads();
  }

  // normalize + scatter to bf16 [B,Sq,DMODEL] (transpose(0,2,1,3).reshape)
  #pragma unroll
  for (int t = 0; t < 4; ++t) {
    int col = h * DHEAD + t * 16 + nl;
    #pragma unroll
    for (int e = 0; e < 8; ++e) {
      int qg = q0 + qr0 + mhalf + e;
      float denom = l_i[e];
      float o = (denom > 0.f) ? Of[t][e] / denom : 0.f;   // nan_to_num(softmax)
      Out[((size_t)b * Sq + qg) * DMODEL + col] = f2bf_bits(o);
    }
  }
}

// ---------------- fused residual-add + LayerNorm (f32 + bf16 outputs) ----------------
__global__ __launch_bounds__(256) void add_ln_kernel(
    const float* __restrict__ X, const float* __restrict__ R,
    const float* __restrict__ g, const float* __restrict__ bta,
    float* __restrict__ Out, unsigned short* __restrict__ OutB) {
  int lane = threadIdx.x & 31;
  int wave = threadIdx.x >> 5;
  size_t row = (size_t)blockIdx.x * 8 + wave;
  const float* px = X + row * DMODEL;
  const float* pr = R + row * DMODEL;
  float v[16];
  float s = 0.f;
  #pragma unroll
  for (int i = 0; i < 16; ++i) {
    int c = lane + i * 32;
    v[i] = px[c] + pr[c];
    s += v[i];
  }
  #pragma unroll
  for (int off = 1; off < 32; off <<= 1) s += __shfl_xor(s, off, 32);
  float mean = s * (1.f / DMODEL);
  float vs = 0.f;
  #pragma unroll
  for (int i = 0; i < 16; ++i) { float d = v[i] - mean; vs += d * d; }
  #pragma unroll
  for (int off = 1; off < 32; off <<= 1) vs += __shfl_xor(vs, off, 32);
  float rstd = rsqrtf(vs * (1.f / DMODEL) + 1e-5f);
  #pragma unroll
  for (int i = 0; i < 16; ++i) {
    int c = lane + i * 32;
    float o = (v[i] - mean) * rstd * g[c] + bta[c];
    Out[row * DMODEL + c]  = o;
    OutB[row * DMODEL + c] = f2bf_bits(o);
  }
}

// ---------------- host-side orchestration ----------------
static inline void gemm(hipStream_t st, const unsigned short* A,
                        const unsigned short* Wt, const float* bias,
                        float* Cf, unsigned short* Cb, int M, int N, int K,
                        int relu) {
  dim3 grid(N / 256, M / 64);
  gemm_bf16_kernel<<<grid, 256, 0, st>>>(A, Wt, bias, Cf, Cb, M, N, K, relu);
}
static inline void attn(hipStream_t st, const unsigned short* Q,
                        const unsigned short* K, const unsigned short* V,
                        unsigned short* O, const int* mask, const int* pad) {
  dim3 grid(SEQ / 64, BATCH * NHEAD);
  attention_kernel<<<grid, 128, 0, st>>>(Q, K, V, O, mask, pad, SEQ, SEQ);
}
static inline void add_ln(hipStream_t st, const float* X, const float* R,
                          const float* g, const float* b, float* O,
                          unsigned short* Ob) {
  add_ln_kernel<<<ROWS / 8, 256, 0, st>>>(X, R, g, b, O, Ob);
}
// convert + transpose one weight matrix into the bf16 arena
static inline unsigned short* conv_w(hipStream_t st, const float* W,
                                     unsigned short** cur, int K, int N) {
  unsigned short* dst = *cur;
  *cur += (size_t)K * N;
  dim3 grid(N / 32, K / 32), blk(32, 8);
  transpose_cvt_kernel<<<grid, blk, 0, st>>>(W, dst, K, N);
  return dst;
}

extern "C" void kernel_launch(void* const* d_in, const int* in_sizes, int n_in,
                              void* d_out, int out_size, void* d_ws,
                              size_t ws_size, hipStream_t stream) {
  (void)in_sizes; (void)n_in; (void)out_size; (void)ws_size;
  const int*   src        = (const int*)d_in[0];
  const int*   tgt        = (const int*)d_in[1];
  const int*   src_mask   = (const int*)d_in[2];
  const int*   tgt_mask   = (const int*)d_in[3];
  const int*   src_pad    = (const int*)d_in[4];
  const int*   tgt_pad    = (const int*)d_in[5];
  const int*   cross_pad  = (const int*)d_in[6];
  const float* src_emb    = (const float*)d_in[7];
  const float* tgt_emb    = (const float*)d_in[8];
  const float* pe         = (const float*)d_in[9];
  const float* e_qkv_w    = (const float*)d_in[10];
  const float* e_qkv_b    = (const float*)d_in[11];
  const float* e_o_w      = (const float*)d_in[12];
  const float* e_o_b      = (const float*)d_in[13];
  const float* e_ff1_w    = (const float*)d_in[14];
  const float* e_ff1_b    = (const float*)d_in[15];
  const float* e_ff2_w    = (const float*)d_in[16];
  const float* e_ff2_b    = (const float*)d_in[17];
  const float* e_ln1_g    = (const float*)d_in[18];
  const float* e_ln1_b    = (const float*)d_in[19];
  const float* e_ln2_g    = (const float*)d_in[20];
  const float* e_ln2_b    = (const float*)d_in[21];
  const float* d_sa_qkv_w = (const float*)d_in[22];
  const float* d_sa_qkv_b = (const float*)d_in[23];
  const float* d_sa_o_w   = (const float*)d_in[24];
  const float* d_sa_o_b   = (const float*)d_in[25];
  const float* d_ca_qkv_w = (const float*)d_in[26];
  const float* d_ca_qkv_b = (const float*)d_in[27];
  const float* d_ca_o_w   = (const float*)d_in[28];
  const float* d_ca_o_b   = (const float*)d_in[29];
  const float* d_ff1_w    = (const float*)d_in[30];
  const float* d_ff1_b    = (const float*)d_in[31];
  const float* d_ff2_w    = (const float*)d_in[32];
  const float* d_ff2_b    = (const float*)d_in[33];
  const float* d_ln1_g    = (const float*)d_in[34];
  const float* d_ln1_b    = (const float*)d_in[35];
  const float* d_ln2_g    = (const float*)d_in[36];
  const float* d_ln2_b    = (const float*)d_in[37];
  const float* d_ln3_g    = (const float*)d_in[38];
  const float* d_ln3_b    = (const float*)d_in[39];
  const float* gen_w      = (const float*)d_in[40];
  const float* gen_b      = (const float*)d_in[41];

  const size_t XE = (size_t)ROWS * DMODEL;    // 2M elements
  const size_t DD = (size_t)DMODEL * DMODEL;

  // fp32 activation buffers
  float* x    = (float*)d_ws;
  float* y    = x + XE;
  float* proj = y + XE;
  // bf16 activation buffers
  unsigned short* us    = (unsigned short*)(proj + XE);
  unsigned short* xb    = us; us += XE;
  unsigned short* yb    = us; us += XE;
  unsigned short* qb    = us; us += XE;
  unsigned short* kbuf  = us; us += XE;
  unsigned short* vbuf  = us; us += XE;
  unsigned short* valsb = us; us += XE;
  unsigned short* ffb   = us; us += (size_t)ROWS * FFDIM;

  // ---- one-shot weight conversion to bf16, transposed [N][K] (L2-resident) ----
  unsigned short *ewq[NLAYERS][3], *ewo[NLAYERS], *ewf1[NLAYERS], *ewf2[NLAYERS];
  unsigned short *wsa[NLAYERS][3], *wsao[NLAYERS];
  unsigned short *wca[NLAYERS][3], *wcao[NLAYERS];
  unsigned short *wf1[NLAYERS], *wf2[NLAYERS], *wgen;
  for (int l = 0; l < NLAYERS; ++l) {
    for (int i = 0; i < 3; ++i)
      ewq[l][i] = conv_w(stream, e_qkv_w + ((size_t)l * 3 + i) * DD, &us, DMODEL, DMODEL);
    ewo[l]  = conv_w(stream, e_o_w  + (size_t)l * DD, &us, DMODEL, DMODEL);
    ewf1[l] = conv_w(stream, e_ff1_w + (size_t)l * DMODEL * FFDIM, &us, DMODEL, FFDIM);
    ewf2[l] = conv_w(stream, e_ff2_w + (size_t)l * FFDIM * DMODEL, &us, FFDIM, DMODEL);
  }
  for (int l = 0; l < NLAYERS; ++l) {
    for (int i = 0; i < 3; ++i)
      wsa[l][i] = conv_w(stream, d_sa_qkv_w + ((size_t)l * 3 + i) * DD, &us, DMODEL, DMODEL);
    wsao[l] = conv_w(stream, d_sa_o_w + (size_t)l * DD, &us, DMODEL, DMODEL);
    for (int i = 0; i < 3; ++i)
      wca[l][i] = conv_w(stream, d_ca_qkv_w + ((size_t)l * 3 + i) * DD, &us, DMODEL, DMODEL);
    wcao[l] = conv_w(stream, d_ca_o_w + (size_t)l * DD, &us, DMODEL, DMODEL);
    wf1[l]  = conv_w(stream, d_ff1_w + (size_t)l * DMODEL * FFDIM, &us, DMODEL, FFDIM);
    wf2[l]  = conv_w(stream, d_ff2_w + (size_t)l * FFDIM * DMODEL, &us, FFDIM, DMODEL);
  }
  wgen = conv_w(stream, gen_w, &us, DMODEL, VOCAB);

  // ---------------- encoder ----------------
  embed_kernel<<<ROWS, 256, 0, stream>>>(src, src_emb, pe, x, xb);
  for (int l = 0; l < NLAYERS; ++l) {
    const float* Bqkv = e_qkv_b + (size_t)l * 3 * DMODEL;
    gemm(stream, xb, ewq[l][0], Bqkv + 0 * DMODEL, nullptr, qb,   ROWS, DMODEL, DMODEL, 0);
    gemm(stream, xb, ewq[l][1], Bqkv + 1 * DMODEL, nullptr, kbuf, ROWS, DMODEL, DMODEL, 0);
    gemm(stream, xb, ewq[l][2], Bqkv + 2 * DMODEL, nullptr, vbuf, ROWS, DMODEL, DMODEL, 0);
    attn(stream, qb, kbuf, vbuf, valsb, src_mask, src_pad);
    gemm(stream, valsb, ewo[l], e_o_b + (size_t)l * DMODEL, proj, nullptr, ROWS, DMODEL, DMODEL, 0);
    add_ln(stream, x, proj, e_ln1_g + (size_t)l * DMODEL, e_ln1_b + (size_t)l * DMODEL, x, xb);
    gemm(stream, xb, ewf1[l], e_ff1_b + (size_t)l * FFDIM, nullptr, ffb, ROWS, FFDIM, DMODEL, 1);
    gemm(stream, ffb, ewf2[l], e_ff2_b + (size_t)l * DMODEL, proj, nullptr, ROWS, DMODEL, FFDIM, 0);
    add_ln(stream, x, proj, e_ln2_g + (size_t)l * DMODEL, e_ln2_b + (size_t)l * DMODEL, x, xb);
  }

  // ---------------- decoder ----------------
  embed_kernel<<<ROWS, 256, 0, stream>>>(tgt, tgt_emb, pe, y, yb);
  for (int l = 0; l < NLAYERS; ++l) {
    // causal self-attention
    const float* Bsa = d_sa_qkv_b + (size_t)l * 3 * DMODEL;
    gemm(stream, yb, wsa[l][0], Bsa + 0 * DMODEL, nullptr, qb,   ROWS, DMODEL, DMODEL, 0);
    gemm(stream, yb, wsa[l][1], Bsa + 1 * DMODEL, nullptr, kbuf, ROWS, DMODEL, DMODEL, 0);
    gemm(stream, yb, wsa[l][2], Bsa + 2 * DMODEL, nullptr, vbuf, ROWS, DMODEL, DMODEL, 0);
    attn(stream, qb, kbuf, vbuf, valsb, tgt_mask, tgt_pad);
    gemm(stream, valsb, wsao[l], d_sa_o_b + (size_t)l * DMODEL, proj, nullptr, ROWS, DMODEL, DMODEL, 0);
    add_ln(stream, y, proj, d_ln1_g + (size_t)l * DMODEL, d_ln1_b + (size_t)l * DMODEL, y, yb);
    // cross-attention (K/V from encoder memory xb)
    const float* Bca = d_ca_qkv_b + (size_t)l * 3 * DMODEL;
    gemm(stream, yb, wca[l][0], Bca + 0 * DMODEL, nullptr, qb,   ROWS, DMODEL, DMODEL, 0);
    gemm(stream, xb, wca[l][1], Bca + 1 * DMODEL, nullptr, kbuf, ROWS, DMODEL, DMODEL, 0);
    gemm(stream, xb, wca[l][2], Bca + 2 * DMODEL, nullptr, vbuf, ROWS, DMODEL, DMODEL, 0);
    attn(stream, qb, kbuf, vbuf, valsb, nullptr, cross_pad);
    gemm(stream, valsb, wcao[l], d_ca_o_b + (size_t)l * DMODEL, proj, nullptr, ROWS, DMODEL, DMODEL, 0);
    add_ln(stream, y, proj, d_ln2_g + (size_t)l * DMODEL, d_ln2_b + (size_t)l * DMODEL, y, yb);
    // FFN
    gemm(stream, yb, wf1[l], d_ff1_b + (size_t)l * FFDIM, nullptr, ffb, ROWS, FFDIM, DMODEL, 1);
    gemm(stream, ffb, wf2[l], d_ff2_b + (size_t)l * DMODEL, proj, nullptr, ROWS, DMODEL, FFDIM, 0);
    add_ln(stream, y, proj, d_ln3_g + (size_t)l * DMODEL, d_ln3_b + (size_t)l * DMODEL, y, yb);
  }

  // ---------------- generator ----------------
  gemm(stream, yb, wgen, gen_b, (float*)d_out, nullptr, ROWS, VOCAB, DMODEL, 0);
}
